// DeltaOnly_3204045603527
// MI455X (gfx1250) — matrix operand order repaired and checked by reading, not compiled, not run on hardware
//
#include <hip/hip_runtime.h>
#include <hip/hip_bf16.h>

// H=32, V=64, B=1024, L=512
#define Hh 32
#define Vv 64
#define Bb 1024
#define Ll 512

typedef __attribute__((ext_vector_type(2)))  float    v2f;
typedef __attribute__((ext_vector_type(8)))  float    v8f;
typedef __attribute__((ext_vector_type(16))) _Float16 v16h;

// ---------------------------------------------------------------------------
// Kernel 1: per-vocab encoder table + per-token norms + fused output weights.
// One block, 64 threads: thread t handles vocab token t AND column t of
// W = read_w @ out_w, biasW = read_b @ out_w + out_b.
// ---------------------------------------------------------------------------
__global__ __launch_bounds__(64)
void prep_kernel(const float* __restrict__ embed, const float* __restrict__ w1,
                 const float* __restrict__ b1,    const float* __restrict__ w2,
                 const float* __restrict__ b2,    const float* __restrict__ gamma,
                 const float* __restrict__ beta,  const float* __restrict__ read_w,
                 const float* __restrict__ read_b,const float* __restrict__ out_w,
                 const float* __restrict__ out_b,
                 float* __restrict__ enc, float* __restrict__ kn2,
                 float* __restrict__ invd, float* __restrict__ W,
                 float* __restrict__ biasW)
{
    const int t = threadIdx.x; // 0..63

    // ---- encoder for vocab token t ----
    float h[Hh], x[Hh];
    #pragma unroll
    for (int i = 0; i < Hh; ++i) { h[i] = embed[t * Hh + i]; x[i] = h[i]; }

    // x = h + relu(h@w1 + b1)@w2  (b2 added after)
    for (int j = 0; j < 2 * Hh; ++j) {
        float s = b1[j];
        #pragma unroll
        for (int i = 0; i < Hh; ++i) s = fmaf(h[i], w1[i * (2 * Hh) + j], s);
        s = fmaxf(s, 0.0f);
        #pragma unroll
        for (int i = 0; i < Hh; ++i) x[i] = fmaf(s, w2[j * Hh + i], x[i]);
    }
    #pragma unroll
    for (int i = 0; i < Hh; ++i) x[i] += b2[i];

    // LayerNorm over H
    float mu = 0.0f;
    #pragma unroll
    for (int i = 0; i < Hh; ++i) mu += x[i];
    mu *= (1.0f / Hh);
    float var = 0.0f;
    #pragma unroll
    for (int i = 0; i < Hh; ++i) { float d = x[i] - mu; var = fmaf(d, d, var); }
    var *= (1.0f / Hh);
    const float inv = 1.0f / sqrtf(var + 1e-5f);

    float s2 = 0.0f;
    #pragma unroll
    for (int i = 0; i < Hh; ++i) {
        float e = (x[i] - mu) * inv * gamma[i] + beta[i];
        enc[t * Hh + i] = e;
        s2 = fmaf(e, e, s2);
    }
    kn2[t]  = s2;                    // ||k||^2
    invd[t] = 1.0f / (s2 + 1e-6f);   // 1/denom

    // ---- fused output projection: column t of W = read_w @ out_w ----
    float bw = out_b[t];
    #pragma unroll
    for (int k = 0; k < Hh; ++k) bw = fmaf(read_b[k], out_w[k * Vv + t], bw);
    biasW[t] = bw;
    for (int i = 0; i < Hh; ++i) {
        float s = 0.0f;
        #pragma unroll
        for (int k = 0; k < Hh; ++k) s = fmaf(read_w[i * Hh + k], out_w[k * Vv + t], s);
        W[i * Vv + t] = s;
    }
}

// ---------------------------------------------------------------------------
// Kernel 2: delta-rule memory. One wave32 per batch element; lane i owns row i
// of M (32 VGPRs). enc table + seq rows staged in LDS. Energy gate uses
// delta_E = 2*(u . v_pred) + ||u||^2 * ||k||^2  > 0.
// ---------------------------------------------------------------------------
__global__ __launch_bounds__(256)
void delta_kernel(const int* __restrict__ seq, const float* __restrict__ enc,
                  const float* __restrict__ kn2, const float* __restrict__ invd,
                  float* __restrict__ rout)
{
    __shared__ __align__(16) float s_enc[Vv * Hh];   // 8 KB
    __shared__ float s_kn2[Vv];
    __shared__ float s_invd[Vv];
    __shared__ int   s_seq[8 * Ll];                  // 16 KB

    const int tid  = threadIdx.x;
    const int lane = tid & 31;
    const int wv   = tid >> 5;                 // wave in block (0..7)
    const int b    = blockIdx.x * 8 + wv;      // batch element

    for (int i = tid; i < Vv * Hh; i += 256) s_enc[i] = enc[i];
    if (tid < Vv) { s_kn2[tid] = kn2[tid]; s_invd[tid] = invd[tid]; }
    const int* srow = seq + (size_t)b * Ll;
    for (int i = lane; i < Ll; i += 32) s_seq[wv * Ll + i] = srow[i];
    __syncthreads();

    float M[Hh];
    #pragma unroll
    for (int j = 0; j < Hh; ++j) M[j] = 0.0f;

    for (int t = 0; t < Ll - 1; ++t) {
        const int tok = s_seq[wv * Ll + t];
        const float* kp = &s_enc[tok * Hh];

        float kv[Hh];
        #pragma unroll
        for (int j4 = 0; j4 < 8; ++j4) {            // LDS broadcast, 8x b128
            float4 q = *(const float4*)(kp + j4 * 4);
            kv[j4 * 4 + 0] = q.x; kv[j4 * 4 + 1] = q.y;
            kv[j4 * 4 + 2] = q.z; kv[j4 * 4 + 3] = q.w;
        }

        // v_pred[lane] = M[lane][:] . k   (4 independent FMA chains)
        float vp0 = 0.f, vp1 = 0.f, vp2 = 0.f, vp3 = 0.f;
        #pragma unroll
        for (int j = 0; j < 8; ++j) {
            vp0 = fmaf(M[4 * j + 0], kv[4 * j + 0], vp0);
            vp1 = fmaf(M[4 * j + 1], kv[4 * j + 1], vp1);
            vp2 = fmaf(M[4 * j + 2], kv[4 * j + 2], vp2);
            vp3 = fmaf(M[4 * j + 3], kv[4 * j + 3], vp3);
        }
        const float vp = (vp0 + vp1) + (vp2 + vp3);

        const float ki = kp[lane];
        const float u  = ki - vp * s_invd[tok];

        // gate: 2*(u.vp) + ||u||^2 * ||k||^2 > 0, via wave32 xor-reduction
        float du = u * vp, uu = u * u;
        #pragma unroll
        for (int off = 16; off > 0; off >>= 1) {
            du += __shfl_xor(du, off, 32);
            uu += __shfl_xor(uu, off, 32);
        }
        const float de = 2.0f * du + uu * s_kn2[tok];
        const float g  = (de > 0.0f) ? u : 0.0f;

        #pragma unroll
        for (int j = 0; j < Hh; ++j) M[j] = fmaf(g, kv[j], M[j]);
    }

    // r = M @ enc[seq[b, L-1]]
    const int tok = s_seq[wv * Ll + (Ll - 1)];
    const float* kp = &s_enc[tok * Hh];
    float a0 = 0.f, a1 = 0.f, a2 = 0.f, a3 = 0.f;
    #pragma unroll
    for (int j = 0; j < 8; ++j) {
        a0 = fmaf(M[4 * j + 0], kp[4 * j + 0], a0);
        a1 = fmaf(M[4 * j + 1], kp[4 * j + 1], a1);
        a2 = fmaf(M[4 * j + 2], kp[4 * j + 2], a2);
        a3 = fmaf(M[4 * j + 3], kp[4 * j + 3], a3);
    }
    rout[(size_t)b * Hh + lane] = (a0 + a1) + (a2 + a3);
}

// ---------------------------------------------------------------------------
// Kernel 3: out(1024x64) = r(1024x32) @ W(32x64) + biasW, via WMMA.
// Full-fp32 path: V_WMMA_F32_16X16X4_F32, 8 K-steps per 16x16 tile.
// A 16x4 layout: VGPR0 = A[m][2*hi], VGPR1 = A[m][2*hi+1].
// B 4x16 layout: VGPRv = B[v + 2*hi][n].  C/D: VGPRr = D[r + 8*hi][n].
// ---------------------------------------------------------------------------
__global__ __launch_bounds__(128)
void outproj_kernel(const float* __restrict__ rmat, const float* __restrict__ W,
                    const float* __restrict__ biasW, float* __restrict__ out)
{
    const int lane = threadIdx.x & 31;
    const int ct   = threadIdx.x >> 5;   // col tile 0..3
    const int rt   = blockIdx.x;         // row tile 0..63
    const int m    = lane & 15;          // A row / D col index
    const int hi   = lane >> 4;
    const int n    = lane & 15;

    v8f c = {};
#if __has_builtin(__builtin_amdgcn_wmma_f32_16x16x4_f32)
    #pragma unroll
    for (int kk = 0; kk < 8; ++kk) {
        const int kb = kk * 4 + 2 * hi;
        v2f a, b;
        a[0] = rmat[(rt * 16 + m) * Hh + kb];
        a[1] = rmat[(rt * 16 + m) * Hh + kb + 1];
        b[0] = W[(kb + 0) * Vv + ct * 16 + n];
        b[1] = W[(kb + 1) * Vv + ct * 16 + n];
        // (neg_a, A, neg_b, B, c_mod, C, reuse_a, reuse_b)
        c = __builtin_amdgcn_wmma_f32_16x16x4_f32(false, a, false, b,
                                                  (short)0, c, false, false);
    }
#else
    // Fallback: single K=32 f16 WMMA (codegen-confirmed builtin).
    v16h a, b;
    #pragma unroll
    for (int e = 0; e < 16; ++e) {
        const int ka = e + 16 * hi;                     // A 16x32 f16 layout (contiguous k order)
        a[e] = (_Float16)rmat[(rt * 16 + m) * Hh + ka];
        const int kb = e + 16 * hi;                     // B 32x16 f16 layout
        b[e] = (_Float16)W[kb * Vv + ct * 16 + n];
    }
    c = __builtin_amdgcn_wmma_f32_16x16x32_f16(false, a, false, b,
                                               (short)0, c, false, false);
#endif

    const float bw = biasW[ct * 16 + n];
    #pragma unroll
    for (int ri = 0; ri < 8; ++ri) {
        const int row = rt * 16 + ri + 8 * hi;
        out[row * Vv + ct * 16 + n] = c[ri] + bw;
    }
}

// ---------------------------------------------------------------------------
extern "C" void kernel_launch(void* const* d_in, const int* in_sizes, int n_in,
                              void* d_out, int out_size, void* d_ws, size_t ws_size,
                              hipStream_t stream)
{
    const int*   seq    = (const int*)  d_in[0];
    const float* embed  = (const float*)d_in[1];
    const float* w1     = (const float*)d_in[2];
    const float* b1     = (const float*)d_in[3];
    const float* w2     = (const float*)d_in[4];
    const float* b2     = (const float*)d_in[5];
    const float* gamma  = (const float*)d_in[6];
    const float* beta   = (const float*)d_in[7];
    const float* read_w = (const float*)d_in[8];
    const float* read_b = (const float*)d_in[9];
    const float* out_w  = (const float*)d_in[10];
    const float* out_b  = (const float*)d_in[11];

    float* ws    = (float*)d_ws;
    float* enc   = ws;                 // 64*32
    float* kn2   = enc   + Vv * Hh;    // 64
    float* invd  = kn2   + Vv;         // 64
    float* Wf    = invd  + Vv;         // 32*64
    float* biasW = Wf    + Hh * Vv;    // 64
    float* rmat  = biasW + Vv;         // 1024*32

    prep_kernel<<<1, 64, 0, stream>>>(embed, w1, b1, w2, b2, gamma, beta,
                                      read_w, read_b, out_w, out_b,
                                      enc, kn2, invd, Wf, biasW);

    delta_kernel<<<Bb / 8, 256, 0, stream>>>(seq, enc, kn2, invd, rmat);

    outproj_kernel<<<Bb / 16, 128, 0, stream>>>(rmat, Wf, biasW, (float*)d_out);
}
